// NMT_81174881894580
// MI455X (gfx1250) — matrix operand order repaired and checked by reading, not compiled, not run on hardware
//
#include <hip/hip_runtime.h>

// ---------------------------------------------------------------------------
// NMT seq2seq for MI455X (gfx1250): bf16 WMMA everywhere, f32 accumulate.
//   T=40 B=64 E=512 H=256 (enc dir), dec hidden 512, L=2, VT=32000, TB=2560
// Dominant cost: projection GEMM 2560x32000x512 (~84 GFLOP) -> v_wmma bf16.
// Wave tile is 32(M)x64(N): 2 A-frags x 4 B-frags -> 8 accumulators, so each
// K-step issues 12 VMEM loads per 8 WMMAs (operand reuse on both sides).
// ---------------------------------------------------------------------------

typedef __bf16 bf16x16 __attribute__((ext_vector_type(16)));
typedef __bf16 bf16x8  __attribute__((ext_vector_type(8)));
typedef float  v8f     __attribute__((ext_vector_type(8)));

union FragBF { bf16x16 v; bf16x8 h[2]; };

static __device__ __forceinline__ __bf16 f2bf(float f) {
    unsigned u = __builtin_bit_cast(unsigned, f);
    unsigned r = u + 0x7FFFu + ((u >> 16) & 1u);          // round-nearest-even
    unsigned short hs = (unsigned short)(r >> 16);
    return __builtin_bit_cast(__bf16, hs);
}

static __device__ __forceinline__ float sigf(float x) {
    return 1.0f / (1.0f + __expf(-x));
}

static __device__ __forceinline__ v8f wmma_bf16(const FragBF& a, const FragBF& b,
                                                v8f c) {
    return __builtin_amdgcn_wmma_f32_16x16x32_bf16(
        false, a.v, false, b.v, (short)0, c, false, false);
}

// ---------------------------------------------------------------------------
// f32 -> bf16 conversion (grid-stride)
// ---------------------------------------------------------------------------
__global__ void nmt_cvt_bf16(const float* __restrict__ in,
                             __bf16* __restrict__ out, long n) {
    long i = (long)blockIdx.x * blockDim.x + threadIdx.x;
    long stride = (long)gridDim.x * blockDim.x;
    for (; i < n; i += stride) out[i] = f2bf(in[i]);
}

// ---------------------------------------------------------------------------
// Embedding gather: out[row, :] = bf16(emb[tok[row], :]), E = 512
// ---------------------------------------------------------------------------
__global__ void nmt_gather_embed(const int* __restrict__ tok,
                                 const float* __restrict__ emb,
                                 __bf16* __restrict__ out) {
    int row = blockIdx.x;
    int t = tok[row];
    const float* src = emb + (size_t)t * 512;
    __bf16* dst = out + (size_t)row * 512;
    for (int e = threadIdx.x; e < 512; e += blockDim.x) dst[e] = f2bf(src[e]);
}

// ---------------------------------------------------------------------------
// C[M,N] = A[M,K](bf16,row) * W[N,K](bf16,row, i.e. B^T) + bias[N]  (f32 out)
// Block: 256 thr = 8 waves; block tile 128x128; wave tile 32(M)x64(N).
// Waves: wm = wave&3 (M strip of 32 rows), wn = wave>>2 (N strip of 64 cols).
// CDNA5 16-bit fragment maps:
//   A 16x32: lane(l16) = M row; elems e<8 -> K=k0+8*half+e ; e>=8 -> +16
//   B 32x16: lane(l16) = N col; elems e<8 -> K=k0+8*half+e ; e>=8 -> +16
//   C/D    : lane(l16) = N col; vgpr r    -> M = 8*half + r
// Requires M%128==0, N%128==0, K%32==0 (true for all call sites).
// ---------------------------------------------------------------------------
__global__ __launch_bounds__(256) void nmt_gemm_bias(
        const __bf16* __restrict__ A, const __bf16* __restrict__ W,
        const float* __restrict__ bias, float* __restrict__ C,
        int M, int N, int K) {
    const int wave  = threadIdx.x >> 5;
    const int lane  = threadIdx.x & 31;
    const int halfl = lane >> 4;
    const int l16   = lane & 15;
    const int m0    = blockIdx.y * 128 + (wave & 3) * 32;
    const int n0    = blockIdx.x * 128 + (wave >> 2) * 64;

    v8f acc[2][4] = {};
    const __bf16* arow0 = A + (size_t)(m0 + l16) * K;
    const __bf16* arow1 = A + (size_t)(m0 + 16 + l16) * K;

    for (int k0 = 0; k0 < K; k0 += 32) {
        FragBF fa0, fa1;
        const __bf16* ap0 = arow0 + k0 + 8 * halfl;
        fa0.h[0] = *(const bf16x8*)ap0;
        fa0.h[1] = *(const bf16x8*)(ap0 + 16);
        const __bf16* ap1 = arow1 + k0 + 8 * halfl;
        fa1.h[0] = *(const bf16x8*)ap1;
        fa1.h[1] = *(const bf16x8*)(ap1 + 16);
        if (k0 + 32 < K) {
            __builtin_prefetch(ap0 + 32, 0, 1);
            __builtin_prefetch(ap1 + 32, 0, 1);
        }
#pragma unroll
        for (int j = 0; j < 4; ++j) {
            FragBF fb;
            const __bf16* bp =
                W + (size_t)(n0 + j * 16 + l16) * K + k0 + 8 * halfl;
            fb.h[0] = *(const bf16x8*)bp;
            fb.h[1] = *(const bf16x8*)(bp + 16);
            acc[0][j] = wmma_bf16(fa0, fb, acc[0][j]);
            acc[1][j] = wmma_bf16(fa1, fb, acc[1][j]);
        }
    }
#pragma unroll
    for (int i = 0; i < 2; ++i) {
#pragma unroll
        for (int j = 0; j < 4; ++j) {
            int col = n0 + j * 16 + l16;
            float bv = bias ? bias[col] : 0.0f;
#pragma unroll
            for (int r = 0; r < 8; ++r) {
                int m = m0 + i * 16 + 8 * halfl + r;
                C[(size_t)m * N + col] = acc[i][j][r] + bv;
            }
        }
    }
}

// ---------------------------------------------------------------------------
// Encoder recurrent scan. One block per direction (grid=2), 512 thr (16 waves).
// Per step: g[64,1024] = h[64,256](LDS bf16) @ Whh^T + xg[t] + bhh, then
// LSTM pointwise update. Gate order i,f,g,o (PyTorch).
// Wave tiling: ms = wave&1 (2 M strips of 32 rows), ng = wave>>1 (8 groups of
// 128 cols); 2 passes of (2x4) accumulators; K-loop 256/32 = 8.
// ---------------------------------------------------------------------------
__global__ __launch_bounds__(512) void nmt_enc_scan(
        const float* __restrict__ xg,    // [2, 2560, 1024] dir-major
        const __bf16* __restrict__ Whh,  // [2, 1024, 256]
        const float* __restrict__ bhh,   // [2, 1024]
        __bf16* __restrict__ y,          // [2560, 512] concat output
        float* __restrict__ fin,         // [2, 64, 512]  (h then c)
        float* __restrict__ gbuf,        // [2, 64, 1024] scratch
        float* __restrict__ cbuf) {      // [2, 64, 256]  scratch
    const int dir = blockIdx.x;
    const float* xgd = xg + (size_t)dir * 2560 * 1024;
    const __bf16* W  = Whh + (size_t)dir * 1024 * 256;
    const float* bh  = bhh + dir * 1024;
    float* gb = gbuf + (size_t)dir * 64 * 1024;
    float* cb = cbuf + (size_t)dir * 64 * 256;

    __shared__ __bf16 hsh[64 * 256];  // 32 KB
    for (int i = threadIdx.x; i < 64 * 256; i += 512) {
        hsh[i] = f2bf(0.0f);
        cb[i] = 0.0f;
    }
    __syncthreads();

    const int wave = threadIdx.x >> 5, lane = threadIdx.x & 31;
    const int ms = wave & 1, ng = wave >> 1;
    const int halfl = lane >> 4, l16 = lane & 15;

    for (int s = 0; s < 40; ++s) {
        const int t = dir ? (39 - s) : s;
        const float* xgt = xgd + (size_t)(t * 64) * 1024;

        for (int pass = 0; pass < 2; ++pass) {
            v8f acc[2][4] = {};
            const int nbase = (ng * 8 + pass * 4) * 16;
            for (int k0 = 0; k0 < 256; k0 += 32) {
                FragBF fa0, fa1;
                const __bf16* ap0 = &hsh[(ms * 32 + l16) * 256 + k0 + 8 * halfl];
                fa0.h[0] = *(const bf16x8*)ap0;
                fa0.h[1] = *(const bf16x8*)(ap0 + 16);
                const __bf16* ap1 =
                    &hsh[(ms * 32 + 16 + l16) * 256 + k0 + 8 * halfl];
                fa1.h[0] = *(const bf16x8*)ap1;
                fa1.h[1] = *(const bf16x8*)(ap1 + 16);
#pragma unroll
                for (int j = 0; j < 4; ++j) {
                    FragBF fb;
                    const __bf16* bp =
                        W + (size_t)(nbase + j * 16 + l16) * 256 + k0 + 8 * halfl;
                    fb.h[0] = *(const bf16x8*)bp;
                    fb.h[1] = *(const bf16x8*)(bp + 16);
                    acc[0][j] = wmma_bf16(fa0, fb, acc[0][j]);
                    acc[1][j] = wmma_bf16(fa1, fb, acc[1][j]);
                }
            }
#pragma unroll
            for (int i = 0; i < 2; ++i) {
#pragma unroll
                for (int j = 0; j < 4; ++j) {
                    int col = nbase + j * 16 + l16;
#pragma unroll
                    for (int r = 0; r < 8; ++r) {
                        int m = ms * 32 + i * 16 + 8 * halfl + r;
                        gb[m * 1024 + col] =
                            acc[i][j][r] + xgt[m * 1024 + col] + bh[col];
                    }
                }
            }
        }
        __syncthreads();

        for (int idx = threadIdx.x; idx < 64 * 256; idx += 512) {
            int m = idx >> 8, j = idx & 255;
            const float* gr = gb + m * 1024;
            float gi = sigf(gr[j]);
            float gf = sigf(gr[256 + j]);
            float gg = tanhf(gr[512 + j]);
            float go = sigf(gr[768 + j]);
            float c = gf * cb[idx] + gi * gg;
            float h = go * tanhf(c);
            cb[idx] = c;
            __bf16 hb = f2bf(h);
            hsh[idx] = hb;
            y[(size_t)(t * 64 + m) * 512 + dir * 256 + j] = hb;
            if (s == 39) {
                fin[(size_t)m * 512 + dir * 256 + j] = h;
                fin[(size_t)(64 + m) * 512 + dir * 256 + j] = c;
            }
        }
        __syncthreads();
    }
}

// ---------------------------------------------------------------------------
// Decoder recurrent scan. One block (512 thr), hidden 512, gates 2048.
// h[64,512] bf16 in dynamic LDS (64 KB). ms = wave&1, ng = wave>>1 (8 groups
// of 256 cols); 4 passes of (2x4) accumulators; K-loop 512/32 = 16.
// Initial h,c from encoder finals.
// ---------------------------------------------------------------------------
__global__ __launch_bounds__(512) void nmt_dec_scan(
        const float* __restrict__ xg,    // [2560, 2048]
        const __bf16* __restrict__ W,    // [2048, 512]
        const float* __restrict__ bh,    // [2048]
        __bf16* __restrict__ y,          // [2560, 512]
        const float* __restrict__ fin,   // [2, 64, 512] (h then c)
        float* __restrict__ gb,          // [64, 2048] scratch
        float* __restrict__ cb) {        // [64, 512]  scratch
    extern __shared__ __bf16 hsh[];  // 64*512 bf16 = 64 KB
    for (int i = threadIdx.x; i < 64 * 512; i += 512) {
        hsh[i] = f2bf(fin[i]);
        cb[i] = fin[64 * 512 + i];
    }
    __syncthreads();

    const int wave = threadIdx.x >> 5, lane = threadIdx.x & 31;
    const int ms = wave & 1, ng = wave >> 1;
    const int halfl = lane >> 4, l16 = lane & 15;

    for (int t = 0; t < 40; ++t) {
        const float* xgt = xg + (size_t)(t * 64) * 2048;

        for (int pass = 0; pass < 4; ++pass) {
            v8f acc[2][4] = {};
            const int nbase = (ng * 16 + pass * 4) * 16;
            for (int k0 = 0; k0 < 512; k0 += 32) {
                FragBF fa0, fa1;
                const __bf16* ap0 = &hsh[(ms * 32 + l16) * 512 + k0 + 8 * halfl];
                fa0.h[0] = *(const bf16x8*)ap0;
                fa0.h[1] = *(const bf16x8*)(ap0 + 16);
                const __bf16* ap1 =
                    &hsh[(ms * 32 + 16 + l16) * 512 + k0 + 8 * halfl];
                fa1.h[0] = *(const bf16x8*)ap1;
                fa1.h[1] = *(const bf16x8*)(ap1 + 16);
#pragma unroll
                for (int j = 0; j < 4; ++j) {
                    FragBF fb;
                    const __bf16* bp =
                        W + (size_t)(nbase + j * 16 + l16) * 512 + k0 + 8 * halfl;
                    fb.h[0] = *(const bf16x8*)bp;
                    fb.h[1] = *(const bf16x8*)(bp + 16);
                    acc[0][j] = wmma_bf16(fa0, fb, acc[0][j]);
                    acc[1][j] = wmma_bf16(fa1, fb, acc[1][j]);
                }
            }
#pragma unroll
            for (int i = 0; i < 2; ++i) {
#pragma unroll
                for (int j = 0; j < 4; ++j) {
                    int col = nbase + j * 16 + l16;
#pragma unroll
                    for (int r = 0; r < 8; ++r) {
                        int m = ms * 32 + i * 16 + 8 * halfl + r;
                        gb[m * 2048 + col] =
                            acc[i][j][r] + xgt[m * 2048 + col] + bh[col];
                    }
                }
            }
        }
        __syncthreads();

        for (int idx = threadIdx.x; idx < 64 * 512; idx += 512) {
            int m = idx >> 9, j = idx & 511;
            const float* gr = gb + m * 2048;
            float gi = sigf(gr[j]);
            float gf = sigf(gr[512 + j]);
            float gg = tanhf(gr[1024 + j]);
            float go = sigf(gr[1536 + j]);
            float c = gf * cb[idx] + gi * gg;
            float h = go * tanhf(c);
            cb[idx] = c;
            __bf16 hb = f2bf(h);
            hsh[idx] = hb;
            y[(size_t)(t * 64 + m) * 512 + j] = hb;
        }
        __syncthreads();
    }
}

// ---------------------------------------------------------------------------
extern "C" void kernel_launch(void* const* d_in, const int* in_sizes, int n_in,
                              void* d_out, int out_size, void* d_ws, size_t ws_size,
                              hipStream_t stream) {
    (void)in_sizes; (void)n_in; (void)out_size; (void)ws_size;

    const int*   src_tok = (const int*)d_in[0];
    const int*   tgt_tok = (const int*)d_in[1];
    const float* src_emb = (const float*)d_in[2];
    const float* tgt_emb = (const float*)d_in[3];
    const float* enc_Wih = (const float*)d_in[4];
    const float* enc_Whh = (const float*)d_in[5];
    const float* enc_bih = (const float*)d_in[6];
    const float* enc_bhh = (const float*)d_in[7];
    const float* dec_Wih = (const float*)d_in[8];
    const float* dec_Whh = (const float*)d_in[9];
    const float* dec_bih = (const float*)d_in[10];
    const float* dec_bhh = (const float*)d_in[11];
    const float* lin_W   = (const float*)d_in[12];
    const float* lin_b   = (const float*)d_in[13];
    float* out = (float*)d_out;

    char* ws = (char*)d_ws;
    size_t off = 0;
    auto take = [&](size_t bytes) {
        char* p = ws + off;
        off += (bytes + 255) & ~(size_t)255;
        return p;
    };

    float*  xg   = (float*) take((size_t)2560 * 2048 * 4);   // gate preacts
    __bf16* bufA = (__bf16*)take((size_t)2560 * 512 * 2);    // activation ping
    __bf16* bufB = (__bf16*)take((size_t)2560 * 512 * 2);    // activation pong
    __bf16* bufT = (__bf16*)take((size_t)2560 * 512 * 2);    // tgt embeddings
    float*  gbuf = (float*) take((size_t)2 * 64 * 2048 * 4); // scan gate scratch
    float*  cbuf = (float*) take((size_t)2 * 64 * 512 * 4);  // scan cell scratch
    float*  fin  = (float*) take((size_t)2 * 2 * 64 * 512 * 4); // per-layer finals
    __bf16* wEih = (__bf16*)take((size_t)2 * 2 * 1024 * 512 * 2);
    __bf16* wEhh = (__bf16*)take((size_t)2 * 2 * 1024 * 256 * 2);
    __bf16* wDih = (__bf16*)take((size_t)2 * 2048 * 512 * 2);
    __bf16* wDhh = (__bf16*)take((size_t)2 * 2048 * 1024 * 2);
    __bf16* wLin = (__bf16*)take((size_t)32000 * 512 * 2);

    auto cvt = [&](const float* a, __bf16* b, long n) {
        int grid = (int)((n + 255) / 256);
        if (grid > 4096) grid = 4096;
        nmt_cvt_bf16<<<grid, 256, 0, stream>>>(a, b, n);
    };
    cvt(enc_Wih, wEih, 2L * 2 * 1024 * 512);
    cvt(enc_Whh, wEhh, 2L * 2 * 1024 * 256);
    cvt(dec_Wih, wDih, 2L * 2048 * 512);
    cvt(dec_Whh, wDhh, 2L * 2048 * 1024);
    cvt(lin_W,   wLin, 32000L * 512);

    nmt_gather_embed<<<2560, 256, 0, stream>>>(src_tok, src_emb, bufA);
    nmt_gather_embed<<<2560, 256, 0, stream>>>(tgt_tok, tgt_emb, bufT);

    // ---- encoder: 2 stacked bidirectional layers -------------------------
    __bf16* cur = bufA;
    __bf16* nxt = bufB;
    for (int l = 0; l < 2; ++l) {
        for (int d = 0; d < 2; ++d) {
            nmt_gemm_bias<<<dim3(1024 / 128, 2560 / 128), 256, 0, stream>>>(
                cur, wEih + (size_t)(l * 2 + d) * 1024 * 512,
                enc_bih + (l * 2 + d) * 1024,
                xg + (size_t)d * 2560 * 1024, 2560, 1024, 512);
        }
        nmt_enc_scan<<<2, 512, 0, stream>>>(
            xg, wEhh + (size_t)l * 2 * 1024 * 256, enc_bhh + l * 2 * 1024,
            nxt, fin + (size_t)l * 2 * 64 * 512, gbuf, cbuf);
        __bf16* tswap = cur; cur = nxt; nxt = tswap;
    }
    // encoder output (in `cur`) is not consumed further; only finals are.

    // ---- decoder: 2 stacked layers, hidden 512, init from enc finals ----
    __bf16* dcur = bufT;
    __bf16* dnxt = nxt;     // free buffer
    __bf16* dalt = cur;     // may be overwritten
    for (int l = 0; l < 2; ++l) {
        nmt_gemm_bias<<<dim3(2048 / 128, 2560 / 128), 256, 0, stream>>>(
            dcur, wDih + (size_t)l * 2048 * 512, dec_bih + l * 2048,
            xg, 2560, 2048, 512);
        nmt_dec_scan<<<1, 512, 64 * 512 * sizeof(__bf16), stream>>>(
            xg, wDhh + (size_t)l * 2048 * 1024, dec_bhh + l * 2048,
            dnxt, fin + (size_t)l * 2 * 64 * 512, gbuf, cbuf);
        dcur = dnxt;
        dnxt = (l == 0) ? dalt : bufT;
    }

    // ---- output projection: [2560,512] x [512,32000] + bias -> f32 ------
    nmt_gemm_bias<<<dim3(32000 / 128, 2560 / 128), 256, 0, stream>>>(
        dcur, wLin, lin_b, out, 2560, 32000, 512);
}
